// GNN_3633542333074
// MI455X (gfx1250) — matrix operand order repaired and checked by reading, not compiled, not run on hardware
//
#include <hip/hip_runtime.h>

typedef __attribute__((ext_vector_type(16))) _Float16 v16h;
typedef __attribute__((ext_vector_type(8)))  float    v8f;

static constexpr int BS = 2;
static constexpr int NN = 50000;
static constexpr int NE = 400000;
static constexpr int DN = 128;
static constexpr int DE = 128;

#define WMMA16(A,B,C) __builtin_amdgcn_wmma_f32_16x16x32_f16(false,(A),false,(B),(short)0,(C),false,false)

// ---- monotone float<->uint encode for atomicMax-based segment max ----
__device__ __forceinline__ unsigned fenc(float x){
  unsigned u = __float_as_uint(x);
  return (u & 0x80000000u) ? ~u : (u | 0x80000000u);
}
__device__ __forceinline__ float fdec(unsigned u){
  unsigned b = (u & 0x80000000u) ? (u ^ 0x80000000u) : ~u;
  return __uint_as_float(b);
}

// B fragment: packed 32x16 tile, 512 halfs, lane-major (lane*16 contiguous)
__device__ __forceinline__ v16h loadB(const _Float16* __restrict__ W, int tile, int lane){
  return *(const v16h*)(W + ((size_t)tile << 9) + (lane << 4));
}

template<int NT>
__device__ __forceinline__ void zeroacc(v8f (&acc)[NT]){
  #pragma unroll
  for (int nt=0; nt<NT; ++nt)
    #pragma unroll
    for (int r=0; r<8; ++r) acc[nt][r] = 0.f;
}

template<int NT>
__device__ __forceinline__ void zeroacc2(v8f (&acc)[2][NT]){
  #pragma unroll
  for (int st=0; st<2; ++st) zeroacc(acc[st]);
}

// ---------------- M=16 variants (node kernel) ----------------
template<int KSTEPS, int NT>
__device__ __forceinline__ void layer_lds(const _Float16* Rsrc, const _Float16* __restrict__ W,
                                          int lane, int row, int hi8, v8f (&acc)[NT]){
  const _Float16* rb = Rsrc + row * (KSTEPS*32);
  #pragma unroll
  for (int kk=0; kk<KSTEPS; ++kk){
    v16h a;
    const _Float16* s1 = rb + kk*32 + hi8;
    const _Float16* s2 = rb + kk*32 + 16 + hi8;
    #pragma unroll
    for (int j=0; j<8; ++j){ a[j] = s1[j]; a[j+8] = s2[j]; }
    #pragma unroll
    for (int nt=0; nt<NT; ++nt){
      v16h bf = loadB(W, kk*NT+nt, lane);
      acc[nt] = WMMA16(a, bf, acc[nt]);
    }
  }
}

template<int NT>
__device__ __forceinline__ void silu_to_lds(v8f (&acc)[NT], const float* __restrict__ bias,
                                            _Float16* Rdst, int row, int hi){
  #pragma unroll
  for (int nt=0; nt<NT; ++nt){
    const int n = nt*16 + row;
    const float bv = bias[n];
    #pragma unroll
    for (int r=0; r<8; ++r){
      const int m = r + hi*8;
      float h = acc[nt][r] + bv;
      h = h / (1.0f + __expf(-h));           // SiLU
      Rdst[m*(NT*16)+n] = (_Float16)h;
    }
  }
}

// ---------------- M=32 variants (edge kernel): one B load feeds two WMMAs ----------------
template<int KSTEPS, int NT>
__device__ __forceinline__ void layer_lds2(const _Float16* Rsrc, const _Float16* __restrict__ W,
                                           int lane, int row, int hi8, v8f (&acc)[2][NT]){
  #pragma unroll
  for (int kk=0; kk<KSTEPS; ++kk){
    v16h a[2];
    #pragma unroll
    for (int st=0; st<2; ++st){
      const _Float16* rb = Rsrc + (st*16 + row)*(KSTEPS*32) + kk*32;
      #pragma unroll
      for (int j=0; j<8; ++j){ a[st][j] = rb[hi8+j]; a[st][j+8] = rb[16+hi8+j]; }
    }
    #pragma unroll
    for (int nt=0; nt<NT; ++nt){
      v16h bf = loadB(W, kk*NT+nt, lane);
      acc[0][nt] = WMMA16(a[0], bf, acc[0][nt]);
      acc[1][nt] = WMMA16(a[1], bf, acc[1][nt]);
    }
  }
}

template<int NT>
__device__ __forceinline__ void silu2_to_lds(v8f (&acc)[2][NT], const float* __restrict__ bias,
                                             _Float16* Rdst, int row, int hi){
  #pragma unroll
  for (int st=0; st<2; ++st)
    #pragma unroll
    for (int nt=0; nt<NT; ++nt){
      const int n = nt*16 + row;
      const float bv = bias[n];
      #pragma unroll
      for (int r=0; r<8; ++r){
        const int m = st*16 + r + hi*8;
        float h = acc[st][nt][r] + bv;
        h = h / (1.0f + __expf(-h));
        Rdst[m*(NT*16)+n] = (_Float16)h;
      }
    }
}

// -------- weight packing: f32 (K x N, row-major) -> f16 WMMA B-fragment tiles --------
__global__ __launch_bounds__(256) void pack_weights(const float* __restrict__ W,
                                                    _Float16* __restrict__ P, int K, int Nd){
  int o = blockIdx.x*blockDim.x + threadIdx.x;
  if (o >= K*Nd) return;
  int tile = o >> 9, within = o & 511;
  int lane = within >> 4, j = within & 15;
  int NT = Nd >> 4;
  int kt = tile / NT, nt = tile - kt*NT;
  // 16-bit B 32x16: lanes<16 hold K=0..7,16..23, lanes>=16 hold K=8..15,24..31, N = lane&15
  int k = kt*32 + ((j & 8) << 1) + ((lane & 16) ? 8 : 0) + (j & 7);
  int n = nt*16 + (lane & 15);
  P[o] = (_Float16)W[(size_t)k*Nd + n];
}

// -------- fused edge pipeline: 32 edges per wave --------
struct EdgeParams {
  const float* V; const float* E; const int* edges;
  const _Float16 *pE0,*pE1,*pS0,*pS1,*pR0w,*pR1w,*pAS0,*pAR0;
  const float *bE0,*bE1,*bS0,*bS1,*bR0,*bR1,*bAS0,*bAR0;
  const float *wAS1,*bAS1,*wAR1,*bAR1;
  float* outEdge;
  _Float16 *msgS,*msgR;
  float *logitS,*logitR;
  unsigned *maxS,*maxR;
};

__global__ __launch_bounds__(64) void edge_kernel(EdgeParams p){
  const int lane = threadIdx.x & 31;
  const int wv   = threadIdx.x >> 5;
  const int tile = blockIdx.x*2 + wv;                  // 0..24999, exact
  const int b    = tile / (NE/32);
  const int e0   = (tile - b*(NE/32))*32;
  const int row  = lane & 15;
  const int hi   = lane >> 4;
  const int hi8  = hi*8;

  __shared__ _Float16 lds0[2][4096];                   // per-wave edge_emb tile (32x128)
  __shared__ _Float16 lds1[2][4096];                   // per-wave hidden scratch (32x128)
  _Float16* R0 = lds0[wv];
  _Float16* R1 = lds1[wv];

  int sIdx[2], rIdx[2];
  const float *srow[2], *rrow[2], *erow[2];
  #pragma unroll
  for (int st=0; st<2; ++st){
    const int* ei = p.edges + ((size_t)b*NE + e0 + st*16 + row)*2;
    sIdx[st] = ei[0]; rIdx[st] = ei[1];
    srow[st] = p.V + ((size_t)b*NN + sIdx[st])*DN;
    rrow[st] = p.V + ((size_t)b*NN + rIdx[st])*DN;
    erow[st] = p.E + ((size_t)b*NE + e0 + st*16 + row)*DE;
  }

  v8f acc[2][8];

  // ---- f_edge layer0: [senders|receivers|E] (32x384) @ W0 (384x128), A direct from global
  zeroacc2(acc);
  #pragma unroll
  for (int kk=0; kk<12; ++kk){
    const int kb = kk*32;
    v16h a[2];
    #pragma unroll
    for (int st=0; st<2; ++st){
      const float* src = (kb < 128) ? (srow[st] + kb)
                       : (kb < 256) ? (rrow[st] + kb - 128)
                                    : (erow[st] + kb - 256);
      #pragma unroll
      for (int j=0; j<8; ++j){
        a[st][j]   = (_Float16)src[hi8 + j];
        a[st][j+8] = (_Float16)src[16 + hi8 + j];
      }
    }
    #pragma unroll
    for (int nt=0; nt<8; ++nt){
      v16h bf = loadB(p.pE0, kk*8+nt, lane);
      acc[0][nt] = WMMA16(a[0], bf, acc[0][nt]);
      acc[1][nt] = WMMA16(a[1], bf, acc[1][nt]);
    }
  }
  silu2_to_lds(acc, p.bE0, R1, row, hi);
  __builtin_amdgcn_wave_barrier();

  // ---- f_edge layer1 -> edge_emb (write out, keep f16 copy in LDS)
  {
    v8f acc2[2][8]; zeroacc2(acc2);
    layer_lds2<4,8>(R1, p.pE1, lane, row, hi8, acc2);
    float* oe = p.outEdge + ((size_t)b*NE + e0)*DE;
    #pragma unroll
    for (int st=0; st<2; ++st)
      #pragma unroll
      for (int nt=0; nt<8; ++nt){
        const int n = nt*16 + row;
        const float bv = p.bE1[n];
        #pragma unroll
        for (int r=0; r<8; ++r){
          const int m = st*16 + r + hi*8;
          const float v = acc2[st][nt][r] + bv;
          oe[(size_t)m*DE + n] = v;
          R0[m*128+n] = (_Float16)v;
        }
      }
  }
  __builtin_amdgcn_wave_barrier();

  // ---- msg_sender MLP: 128 -> 128 (SiLU) -> 64
  zeroacc2(acc);
  layer_lds2<4,8>(R0, p.pS0, lane, row, hi8, acc);
  silu2_to_lds(acc, p.bS0, R1, row, hi);
  __builtin_amdgcn_wave_barrier();
  {
    v8f am[2][4]; zeroacc2(am);
    layer_lds2<4,4>(R1, p.pS1, lane, row, hi8, am);
    _Float16* om = p.msgS + ((size_t)b*NE + e0)*64;
    #pragma unroll
    for (int st=0; st<2; ++st)
      #pragma unroll
      for (int nt=0; nt<4; ++nt){
        const int n = nt*16 + row;
        const float bv = p.bS1[n];
        #pragma unroll
        for (int r=0; r<8; ++r){
          const int m = st*16 + r + hi*8;
          om[m*64+n] = (_Float16)(am[st][nt][r] + bv);
        }
      }
  }
  __builtin_amdgcn_wave_barrier();

  // ---- msg_receiver MLP
  zeroacc2(acc);
  layer_lds2<4,8>(R0, p.pR0w, lane, row, hi8, acc);
  silu2_to_lds(acc, p.bR0, R1, row, hi);
  __builtin_amdgcn_wave_barrier();
  {
    v8f am[2][4]; zeroacc2(am);
    layer_lds2<4,4>(R1, p.pR1w, lane, row, hi8, am);
    _Float16* om = p.msgR + ((size_t)b*NE + e0)*64;
    #pragma unroll
    for (int st=0; st<2; ++st)
      #pragma unroll
      for (int nt=0; nt<4; ++nt){
        const int n = nt*16 + row;
        const float bv = p.bR1[n];
        #pragma unroll
        for (int r=0; r<8; ++r){
          const int m = st*16 + r + hi*8;
          om[m*64+n] = (_Float16)(am[st][nt][r] + bv);
        }
      }
  }
  __builtin_amdgcn_wave_barrier();

  // ---- attn_sender: 128 -> 32 (WMMA, SiLU) -> 1 (VALU dot), clip, atomic seg-max
  {
    v8f aa[2][2]; zeroacc2(aa);
    layer_lds2<4,2>(R0, p.pAS0, lane, row, hi8, aa);
    silu2_to_lds(aa, p.bAS0, R1, row, hi);           // 32x32 tile in R1
    __builtin_amdgcn_wave_barrier();
    float lg = p.bAS1[0];                             // lane = edge row (0..31)
    #pragma unroll
    for (int k=0; k<32; ++k) lg += (float)R1[lane*32+k] * p.wAS1[k];
    lg = fminf(30.0f, fmaxf(-30.0f, lg));
    p.logitS[(size_t)b*NE + e0 + lane] = lg;
    const int si = (lane < 16) ? sIdx[0] : sIdx[1];
    atomicMax(p.maxS + (size_t)b*NN + si, fenc(lg));
  }
  __builtin_amdgcn_wave_barrier();

  // ---- attn_receiver
  {
    v8f aa[2][2]; zeroacc2(aa);
    layer_lds2<4,2>(R0, p.pAR0, lane, row, hi8, aa);
    silu2_to_lds(aa, p.bAR0, R1, row, hi);
    __builtin_amdgcn_wave_barrier();
    float lg = p.bAR1[0];
    #pragma unroll
    for (int k=0; k<32; ++k) lg += (float)R1[lane*32+k] * p.wAR1[k];
    lg = fminf(30.0f, fmaxf(-30.0f, lg));
    p.logitR[(size_t)b*NE + e0 + lane] = lg;
    const int ri = (lane < 16) ? rIdx[0] : rIdx[1];
    atomicMax(p.maxR + (size_t)b*NN + ri, fenc(lg));
  }
}

// -------- pass 2: e = exp(logit - segmax); atomic segsum and e*msg scatter --------
__global__ __launch_bounds__(256) void scatter_kernel(const int* __restrict__ edges,
    const float* __restrict__ logit, const unsigned* __restrict__ segMax,
    float* __restrict__ segSum, const _Float16* __restrict__ msg,
    float* __restrict__ agg, int side){
  long long i = (long long)blockIdx.x*blockDim.x + threadIdx.x;   // 0..BS*NE-1
  if (i >= (long long)BS*NE) return;
  const int b = (int)(i / NE);
  const int idx = edges[i*2 + side];
  const float m = fdec(segMax[(size_t)b*NN + idx]);
  const float ev = __expf(logit[i] - m);
  atomicAdd(segSum + (size_t)b*NN + idx, ev);
  const _Float16* mp = msg + (size_t)i*64;
  float* ap = agg + ((size_t)b*NN + idx)*128 + side*64;
  #pragma unroll 8
  for (int f=0; f<64; ++f) atomicAdd(ap + f, ev * (float)mp[f]);
}

// -------- node MLP --------
struct NodeParams {
  const float* V; const float* agg; const float* sumS; const float* sumR;
  const _Float16 *pN0,*pN1; const float *bN0,*bN1;
  float* outNode;
};

__global__ __launch_bounds__(64) void node_kernel(NodeParams p){
  const int lane = threadIdx.x & 31;
  const int wv   = threadIdx.x >> 5;
  const int tile = blockIdx.x*2 + wv;                 // 0..6249, exact
  const int b    = tile / (NN/16);
  const int n0   = (tile - b*(NN/16))*16;
  const int row  = lane & 15;
  const int hi   = lane >> 4;
  const int hi8  = hi*8;
  const int node = n0 + row;

  __shared__ _Float16 lds[2][2048];
  _Float16* R1 = lds[wv];

  const float* vrow = p.V   + ((size_t)b*NN + node)*DN;
  const float* arow = p.agg + ((size_t)b*NN + node)*128;
  const float s0 = p.sumS[(size_t)b*NN + node];
  const float s1 = p.sumR[(size_t)b*NN + node];
  const float sc0 = (s0 > 0.f) ? (1.f/s0) : 0.f;      // nodes with no edges -> agg 0
  const float sc1 = (s1 > 0.f) ? (1.f/s1) : 0.f;

  v8f acc[8]; zeroacc(acc);
  #pragma unroll
  for (int kk=0; kk<8; ++kk){                          // K = 256: [V | agg0/s | agg1/s]
    const int kb = kk*32;
    const float* src; float sc;
    if (kb < 128){ src = vrow + kb; sc = 1.f; }
    else { const int o = kb - 128; src = arow + o; sc = (o < 64) ? sc0 : sc1; }
    v16h a;
    #pragma unroll
    for (int j=0; j<8; ++j){
      a[j]   = (_Float16)(src[hi8 + j] * sc);
      a[j+8] = (_Float16)(src[16 + hi8 + j] * sc);
    }
    #pragma unroll
    for (int nt=0; nt<8; ++nt){
      v16h bf = loadB(p.pN0, kk*8+nt, lane);
      acc[nt] = WMMA16(a, bf, acc[nt]);
    }
  }
  silu_to_lds(acc, p.bN0, R1, row, hi);
  __builtin_amdgcn_wave_barrier();

  v8f acc2[8]; zeroacc(acc2);
  layer_lds<4,8>(R1, p.pN1, lane, row, hi8, acc2);
  float* on = p.outNode + ((size_t)b*NN + n0)*DN;
  #pragma unroll
  for (int nt=0; nt<8; ++nt){
    const int n = nt*16 + row;
    const float bv = p.bN1[n];
    #pragma unroll
    for (int r=0; r<8; ++r){
      const int m = r + hi*8;
      on[(size_t)m*DN + n] = acc2[nt][r] + bv;
    }
  }
}

extern "C" void kernel_launch(void* const* d_in, const int* in_sizes, int n_in,
                              void* d_out, int out_size, void* d_ws, size_t ws_size,
                              hipStream_t stream){
  (void)n_in; (void)out_size; (void)ws_size;
  // jax pytree flatten sorts dict keys -> E first (102.4M elems); insertion order -> V first.
  const bool sorted = (in_sizes[0] == BS*NE*DE);
  int iV,iE,iEd, ife0w,ife0b,ife1w,ife1b, ims0w,ims0b,ims1w,ims1b,
      imr0w,imr0b,imr1w,imr1b, ias0w,ias0b,ias1w,ias1b,
      iar0w,iar0b,iar1w,iar1b, ifn0w,ifn0b,ifn1w,ifn1b;
  if (!sorted){
    iV=0; iE=1;
    ife0w=2;  ife0b=3;  ife1w=4;  ife1b=5;
    ims0w=6;  ims0b=7;  ims1w=8;  ims1b=9;
    imr0w=10; imr0b=11; imr1w=12; imr1b=13;
    ias0w=14; ias0b=15; ias1w=16; ias1b=17;
    iar0w=18; iar0b=19; iar1w=20; iar1b=21;
    ifn0w=22; ifn0b=23; ifn1w=24; ifn1b=25;
    iEd=26;
  } else {
    iE=0; iV=1; iEd=2;
    iar0b=3;  iar0w=4;  iar1b=5;  iar1w=6;
    ias0b=7;  ias0w=8;  ias1b=9;  ias1w=10;
    ife0b=11; ife0w=12; ife1b=13; ife1w=14;
    imr0b=15; imr0w=16; imr1b=17; imr1w=18;
    ims0b=19; ims0w=20; ims1b=21; ims1w=22;
    ifn0b=23; ifn0w=24; ifn1b=25; ifn1w=26;
  }

  char* ws = (char*)d_ws;
  size_t off = 0;
  auto alloc = [&](size_t bytes)->size_t{
    size_t o = off; off = (off + bytes + 255) & ~(size_t)255; return o;
  };
  const size_t oPE0 = alloc(384*128*2), oPE1 = alloc(128*128*2),
               oPS0 = alloc(128*128*2), oPS1 = alloc(128*64*2),
               oPR0 = alloc(128*128*2), oPR1 = alloc(128*64*2),
               oPA0 = alloc(128*32*2),  oPB0 = alloc(128*32*2),
               oPN0 = alloc(256*128*2), oPN1 = alloc(128*128*2);
  const size_t oMsgS = alloc((size_t)BS*NE*64*2);
  const size_t oMsgR = alloc((size_t)BS*NE*64*2);
  const size_t oLgS  = alloc((size_t)BS*NE*4);
  const size_t oLgR  = alloc((size_t)BS*NE*4);
  const size_t zeroStart = off;
  const size_t oMxS = alloc((size_t)BS*NN*4);
  const size_t oMxR = alloc((size_t)BS*NN*4);
  const size_t oSmS = alloc((size_t)BS*NN*4);
  const size_t oSmR = alloc((size_t)BS*NN*4);
  const size_t oAgg = alloc((size_t)BS*NN*128*4);
  hipMemsetAsync(ws + zeroStart, 0, off - zeroStart, stream);  // segmax 0 sorts below fenc(-inf), sums/agg=0

  auto pk = [&](int wi, size_t o, int K, int Nd){
    pack_weights<<<dim3((unsigned)((K*Nd+255)/256)), dim3(256), 0, stream>>>(
        (const float*)d_in[wi], (_Float16*)(ws+o), K, Nd);
  };
  pk(ife0w,oPE0,384,128); pk(ife1w,oPE1,128,128);
  pk(ims0w,oPS0,128,128); pk(ims1w,oPS1,128,64);
  pk(imr0w,oPR0,128,128); pk(imr1w,oPR1,128,64);
  pk(ias0w,oPA0,128,32);  pk(iar0w,oPB0,128,32);
  pk(ifn0w,oPN0,256,128); pk(ifn1w,oPN1,128,128);

  EdgeParams ep;
  ep.V = (const float*)d_in[iV]; ep.E = (const float*)d_in[iE]; ep.edges = (const int*)d_in[iEd];
  ep.pE0 = (const _Float16*)(ws+oPE0); ep.pE1 = (const _Float16*)(ws+oPE1);
  ep.pS0 = (const _Float16*)(ws+oPS0); ep.pS1 = (const _Float16*)(ws+oPS1);
  ep.pR0w= (const _Float16*)(ws+oPR0); ep.pR1w= (const _Float16*)(ws+oPR1);
  ep.pAS0= (const _Float16*)(ws+oPA0); ep.pAR0= (const _Float16*)(ws+oPB0);
  ep.bE0 = (const float*)d_in[ife0b]; ep.bE1 = (const float*)d_in[ife1b];
  ep.bS0 = (const float*)d_in[ims0b]; ep.bS1 = (const float*)d_in[ims1b];
  ep.bR0 = (const float*)d_in[imr0b]; ep.bR1 = (const float*)d_in[imr1b];
  ep.bAS0= (const float*)d_in[ias0b]; ep.bAR0= (const float*)d_in[iar0b];
  ep.wAS1= (const float*)d_in[ias1w]; ep.bAS1= (const float*)d_in[ias1b];
  ep.wAR1= (const float*)d_in[iar1w]; ep.bAR1= (const float*)d_in[iar1b];
  ep.outEdge = (float*)d_out + (size_t)BS*NN*DN;    // tuple: node_emb first, edge_emb second
  ep.msgS = (_Float16*)(ws+oMsgS); ep.msgR = (_Float16*)(ws+oMsgR);
  ep.logitS = (float*)(ws+oLgS);   ep.logitR = (float*)(ws+oLgR);
  ep.maxS = (unsigned*)(ws+oMxS);  ep.maxR = (unsigned*)(ws+oMxR);
  edge_kernel<<<dim3(12500), dim3(64), 0, stream>>>(ep);       // 25000 tiles of 32 edges / 2 waves

  scatter_kernel<<<dim3(3125), dim3(256), 0, stream>>>(
      (const int*)d_in[iEd], (const float*)(ws+oLgS), (const unsigned*)(ws+oMxS),
      (float*)(ws+oSmS), (const _Float16*)(ws+oMsgS), (float*)(ws+oAgg), 0);
  scatter_kernel<<<dim3(3125), dim3(256), 0, stream>>>(
      (const int*)d_in[iEd], (const float*)(ws+oLgR), (const unsigned*)(ws+oMxR),
      (float*)(ws+oSmR), (const _Float16*)(ws+oMsgR), (float*)(ws+oAgg), 1);

  NodeParams np;
  np.V = (const float*)d_in[iV]; np.agg = (const float*)(ws+oAgg);
  np.sumS = (const float*)(ws+oSmS); np.sumR = (const float*)(ws+oSmR);
  np.pN0 = (const _Float16*)(ws+oPN0); np.pN1 = (const _Float16*)(ws+oPN1);
  np.bN0 = (const float*)d_in[ifn0b]; np.bN1 = (const float*)d_in[ifn1b];
  np.outNode = (float*)d_out;
  node_kernel<<<dim3(3125), dim3(64), 0, stream>>>(np);        // 6250 tiles / 2 waves
}
